// Fire_model_77713138254193
// MI455X (gfx1250) — matrix operand order repaired and checked by reading, not compiled, not run on hardware
//
#include <hip/hip_runtime.h>
#include <hip/hip_bf16.h>

typedef _Float16 v16h __attribute__((ext_vector_type(16)));
typedef _Float16 v8h  __attribute__((ext_vector_type(8)));
typedef float    v8f  __attribute__((ext_vector_type(8)));

// explicit global-address-space views (keeps global_load_b128 after laundering)
typedef __attribute__((address_space(1))) const _Float16 gas_f16;
typedef __attribute__((address_space(1))) const v16h    gas_v16h;

namespace {
constexpr int B_   = 256;
constexpr int T_   = 365;
constexpr int D_   = 16;
constexpr int N_   = 32;
constexpr int DN   = D_ * N_;       // 512
constexpr int KIN  = D_ * (N_ + 1); // 528
constexpr int KPAD = 544;           // 17 * 32
constexpr int KC   = KPAD / 32;     // 17 k-chunks
constexpr int NTG  = 3 * (DN / 16); // 96 n-tiles across i,f,o
constexpr long GP_HALFS = (long)NTG * KC * 32 * 16; // 835584 halfs (1.67 MB)
constexpr long JP_HALFS = (long)D_ * 2 * 32 * 16;   // 16384 halfs  (32 KB)
}

// ---------------- weight packing into WMMA-B lane layout -----------------
// B(32x16 f16) layout: lane = 16*((k/8)%2) + n, half index = k%8 + 8*(k/16).
// Packed so each lane reads 16 contiguous halfs (32B) per (tile, k-chunk).
__global__ void fire_pack(const float* __restrict__ Wi, const float* __restrict__ Wf,
                          const float* __restrict__ Wo, const float* __restrict__ Wj,
                          _Float16* __restrict__ gpack, _Float16* __restrict__ jpack) {
  long idx = (long)blockIdx.x * blockDim.x + threadIdx.x;
  if (idx < GP_HALFS) {
    int h  = idx & 15;
    int l  = (idx >> 4) & 31;
    int kc = (int)((idx >> 9) % KC);
    int nt = (int)(idx / (512 * KC));
    int g  = nt >> 5;                      // 0=i 1=f 2=o
    int row = (nt & 31) * 16 + (l & 15);   // output unit (n)
    int col = kc * 32 + (l >> 4) * 8 + (h & 7) + (h >> 3) * 16; // input (k), zero-padded past 528
    const float* W = (g == 0) ? Wi : (g == 1) ? Wf : Wo;
    float v = (col < KIN) ? W[row * KIN + col] : 0.f;
    gpack[idx] = (_Float16)v;
  } else if (idx < GP_HALFS + JP_HALFS) {
    long j = idx - GP_HALFS;
    int h   = j & 15;
    int l   = (j >> 4) & 31;
    int nt2 = (j >> 9) & 1;
    int d   = (int)(j >> 10);
    int k   = (l >> 4) * 8 + (h & 7) + (h >> 3) * 16;
    int n   = nt2 * 16 + (l & 15);
    jpack[j] = (_Float16)Wj[(d * 32 + k) * 32 + n];
  }
}

// ---------------- main persistent recurrent kernel -----------------------
__device__ __forceinline__ float sigmoidf_(float x) { return 1.f / (1.f + __expf(-x)); }

// A(16x32 f16) layout: lane holds row M=l%16; halfs 0..7 = K base..base+7,
// halfs 8..15 = K base+16..base+23, base = 8*(l/16). Two ds_load_b128.
__device__ __forceinline__ v16h make_a(const _Float16* lo, const _Float16* hi) {
  v8h a = *(const v8h*)lo;
  v8h b = *(const v8h*)hi;
  v16h r;
#pragma unroll
  for (int i = 0; i < 8; ++i) { r[i] = a[i]; r[i + 8] = b[i]; }
  return r;
}

__global__ __launch_bounds__(256, 1)
void fire_main(const float* __restrict__ x,
               const float* __restrict__ u,   // U_j[:,0,:] (D,N)
               const float* __restrict__ bj,
               const float* __restrict__ Wib, const float* __restrict__ Wfb,
               const float* __restrict__ Wob,
               const float* __restrict__ Fa,  const float* __restrict__ Fab,
               const float* __restrict__ Fbw, const float* __restrict__ Fbb,
               const float* __restrict__ Phw, const float* __restrict__ Phb,
               const _Float16* __restrict__ gpack, const _Float16* __restrict__ jpack,
               float* __restrict__ out_mean, float* __restrict__ out_alpha,
               float* __restrict__ out_beta) {
  __shared__ __align__(32) _Float16 inpA[16 * KPAD];   // [x_t | h | zero-pad] per row
  __shared__ __align__(32) _Float16 gact[4][16 * DN];  // sig(i),sig(f),sig(o),tanh(j)
  __shared__ float xf[16 * 16];
  __shared__ float wsum[16 * DN];   // streaming  sum_t e*h
  __shared__ float cL[16 * DN];     // cell state
  __shared__ float biasL[3 * DN];
  __shared__ float uL[DN], bjL[DN], FaL[DN];
  __shared__ float FabL[16];
  __shared__ float muL[256], beL[256], sbL[16];

  const int tid  = threadIdx.x;
  const int w    = tid >> 5;  // wave id (8 waves)
  const int l    = tid & 31;
  const int lhi  = l >> 4;
  const int lm   = l & 15;
  const int bBase = blockIdx.x * 16;
  const int mrow = tid >> 4;  // scalar-phase row
  const int dcol = tid & 15;  // scalar-phase d

  for (int i = tid; i < 16 * KPAD; i += 256) inpA[i] = (_Float16)0.f; // h0=0 + K pad
  for (int i = tid; i < 16 * DN; i += 256) { wsum[i] = 0.f; cL[i] = 0.f; }
  for (int i = tid; i < DN; i += 256) {
    uL[i] = u[i]; bjL[i] = bj[i]; FaL[i] = Fa[i];
    biasL[i] = Wib[i]; biasL[DN + i] = Wfb[i]; biasL[2 * DN + i] = Wob[i];
  }
  if (tid < 16) FabL[tid] = Fab[tid];
  float seAcc = 0.f;  // per-(row,d) running sum of e
  __syncthreads();

  for (int t = 0; t < T_; ++t) {
    // launder base addresses as integers (defeats hoist-to-scratch LICM),
    // then rebuild as addrspace(1) pointers so loads stay global_load_b128.
    unsigned long long ga = (unsigned long long)gpack;
    unsigned long long ja = (unsigned long long)jpack;
    asm volatile("" : "+s"(ga), "+s"(ja));
    gas_f16* gpk = (gas_f16*)ga;
    gas_f16* jpk = (gas_f16*)ja;

    // ---- phase 1: stage x_t (cols 0..15 of inp) ----
    {
      float xv = x[((size_t)(bBase + mrow) * T_ + t) * D_ + dcol];
      xf[mrow * 16 + dcol] = xv;
      inpA[mrow * KPAD + dcol] = (_Float16)xv;
    }
    __syncthreads();

    // ---- phase 2: gate GEMMs, one gate at a time (4 accs live) ----
    for (int g3 = 0; g3 < 3; ++g3) {
      v8f acc[4];
#pragma unroll
      for (int ut = 0; ut < 4; ++ut) {
        float b = biasL[g3 * DN + (4 * w + ut) * 16 + lm];
#pragma unroll
        for (int r = 0; r < 8; ++r) acc[ut][r] = b;
      }
      for (int kc = 0; kc < KC; ++kc) {
        const _Float16* arow = &inpA[lm * KPAD + kc * 32 + 8 * lhi];
        v16h A = make_a(arow, arow + 16);
        if (kc + 1 < KC)  // prefetch next k-chunk via known-global kernarg ptr
          __builtin_prefetch(gpack + (((size_t)(g3 * 32 + 4 * w) * KC + kc + 1) * 32 + l) * 16, 0, 3);
        v16h Bv[4];
#pragma unroll
        for (int ut = 0; ut < 4; ++ut)
          Bv[ut] = *(gas_v16h*)(gpk + (((size_t)(g3 * 32 + 4 * w + ut) * KC + kc) * 32 + l) * 16);
#pragma unroll
        for (int ut = 0; ut < 4; ++ut)
          acc[ut] = __builtin_amdgcn_wmma_f32_16x16x32_f16(
              false, A, false, Bv[ut], (short)0, acc[ut], false, false);
      }
#pragma unroll
      for (int ut = 0; ut < 4; ++ut) {
        int ug = (4 * w + ut) * 16 + lm;
#pragma unroll
        for (int r = 0; r < 8; ++r)
          gact[g3][(r + 8 * lhi) * DN + ug] = (_Float16)sigmoidf_(acc[ut][r]);
      }
    }
    // ---- phase 2b: j_t = tanh(h @ W_j[d] + x_t*u + b_j), d in {2w,2w+1} ----
    {
      v8f accJ[4];
#pragma unroll
      for (int ut = 0; ut < 4; ++ut) {
        int d = 2 * w + (ut >> 1);
        int nloc = (ut & 1) * 16 + lm;
#pragma unroll
        for (int r = 0; r < 8; ++r)
          accJ[ut][r] = xf[(r + 8 * lhi) * 16 + d] * uL[d * 32 + nloc] + bjL[d * 32 + nloc];
      }
#pragma unroll
      for (int dd = 0; dd < 2; ++dd) {
        int d = 2 * w + dd;
        const _Float16* arow = &inpA[lm * KPAD + 16 + d * 32 + 8 * lhi];
        v16h Aj = make_a(arow, arow + 16);
#pragma unroll
        for (int nt2 = 0; nt2 < 2; ++nt2) {
          int ut = dd * 2 + nt2;
          v16h Bv = *(gas_v16h*)(jpk + ((size_t)(d * 2 + nt2) * 32 + l) * 16);
          accJ[ut] = __builtin_amdgcn_wmma_f32_16x16x32_f16(
              false, Aj, false, Bv, (short)0, accJ[ut], false, false);
        }
      }
#pragma unroll
      for (int ut = 0; ut < 4; ++ut) {
        int ug = (2 * w + (ut >> 1)) * 32 + (ut & 1) * 16 + lm;
#pragma unroll
        for (int r = 0; r < 8; ++r)
          gact[3][(r + 8 * lhi) * DN + ug] = (_Float16)tanhf(accJ[ut][r]);
      }
    }
    __syncthreads();  // gates visible; all inpA h reads done

    // ---- phase 3+4 fused: c/h update + streaming alpha, thread=(row,d) ----
    {
      const _Float16* gi = &gact[0][mrow * DN + dcol * 32];
      const _Float16* gf = &gact[1][mrow * DN + dcol * 32];
      const _Float16* go = &gact[2][mrow * DN + dcol * 32];
      const _Float16* gj = &gact[3][mrow * DN + dcol * 32];
      _Float16* hp = &inpA[mrow * KPAD + 16 + dcol * 32];
      float* cp = &cL[mrow * DN + dcol * 32];
      float* wp = &wsum[mrow * DN + dcol * 32];
      float hv[32];
      float s = 0.f;
#pragma unroll
      for (int n = 0; n < 32; ++n) {
        float c = cp[n] * (float)gf[n] + (float)gi[n] * (float)gj[n];
        cp[n] = c;
        float h = (float)go[n] * tanhf(c);
        hv[n] = h;
        hp[n] = (_Float16)h;
        s += h * FaL[dcol * 32 + n];
      }
      float e = __expf(tanhf(s + FabL[dcol]));
      out_alpha[((size_t)(bBase + mrow) * T_ + t) * D_ + dcol] = e; // unnormalized
      seAcc += e;
#pragma unroll
      for (int n = 0; n < 32; ++n) wp[n] += e * hv[n];
    }
    __syncthreads();
  }

  // ---- finalize: g_n, hg, mu, betas, mean ----
  {
    float invSE = 1.f / seAcc;
    const _Float16* hp = &inpA[mrow * KPAD + 16 + dcol * 32]; // h_f
    const float* wp = &wsum[mrow * DN + dcol * 32];
    float mu = Phb[0];
    float bp = Fbb[0];
#pragma unroll
    for (int n = 0; n < 32; ++n) {
      float g  = wp[n] * invSE;   // g_n
      float hf = (float)hp[n];
      mu += g * Phw[n] + hf * Phw[32 + n];
      bp += g * Fbw[n] + hf * Fbw[32 + n];
    }
    muL[tid] = mu;
    beL[tid] = __expf(tanhf(bp));
  }
  __syncthreads();
  if (tid < 16) {
    float s = 0.f;
    for (int d = 0; d < 16; ++d) s += beL[tid * 16 + d];
    sbL[tid] = s;
    float mean = 0.f;
    for (int d = 0; d < 16; ++d) mean += (beL[tid * 16 + d] / s) * muL[tid * 16 + d];
    out_mean[bBase + tid] = mean;
  }
  __syncthreads();
  {
    out_beta[(bBase + mrow) * 16 + dcol] = beL[tid] / sbL[mrow];
    float invSE = 1.f / seAcc;
    for (int t = 0; t < T_; ++t) { // rescale own previously-written e values
      size_t idx = ((size_t)(bBase + mrow) * T_ + t) * D_ + dcol;
      out_alpha[idx] *= invSE;
    }
  }
}

extern "C" void kernel_launch(void* const* d_in, const int* in_sizes, int n_in,
                              void* d_out, int out_size, void* d_ws, size_t ws_size,
                              hipStream_t stream) {
  const float* x   = (const float*)d_in[0];
  const float* Uj  = (const float*)d_in[1];
  const float* Wj  = (const float*)d_in[2];
  const float* bj  = (const float*)d_in[3];
  const float* Wiw = (const float*)d_in[4];
  const float* Wib = (const float*)d_in[5];
  const float* Wfw = (const float*)d_in[6];
  const float* Wfb = (const float*)d_in[7];
  const float* Wow = (const float*)d_in[8];
  const float* Wob = (const float*)d_in[9];
  const float* Fa  = (const float*)d_in[10];
  const float* Fab = (const float*)d_in[11];
  const float* Fbw = (const float*)d_in[12];
  const float* Fbb = (const float*)d_in[13];
  const float* Phw = (const float*)d_in[14];
  const float* Phb = (const float*)d_in[15];

  _Float16* gpack = (_Float16*)d_ws;
  _Float16* jpack = gpack + GP_HALFS;

  float* out_mean  = (float*)d_out;
  float* out_alpha = out_mean + B_;
  float* out_beta  = out_alpha + (size_t)B_ * T_ * D_;

  long total = GP_HALFS + JP_HALFS;
  fire_pack<<<(int)((total + 255) / 256), 256, 0, stream>>>(Wiw, Wfw, Wow, Wj, gpack, jpack);
  fire_main<<<B_ / 16, 256, 0, stream>>>(x, Uj, bj, Wib, Wfb, Wob, Fa, Fab, Fbw, Fbb,
                                         Phw, Phb, gpack, jpack,
                                         out_mean, out_alpha, out_beta);
}
